// IARMEnhanced_36653250904539
// MI455X (gfx1250) — matrix operand order repaired and checked by reading, not compiled, not run on hardware
//
#include <hip/hip_runtime.h>
#include <hip/hip_bf16.h>

#define ALPHA 0.2f
#define EPSLN 1e-5f

#define Bx   8
#define Nx   256
#define INx  256
#define OUTx 256
#define Hx   4
#define HDx  64

typedef __bf16 bf16;
typedef __attribute__((ext_vector_type(16))) __bf16 v16bf;
typedef __attribute__((ext_vector_type(8)))  __bf16 v8bf;
typedef __attribute__((ext_vector_type(8)))  float  v8f;

static __device__ inline v8f wmma_bf16(v16bf a, v16bf b, v8f c) {
  // D = A(16x32 bf16) * B(32x16 bf16) + C(16x16 f32)
  return __builtin_amdgcn_wmma_f32_16x16x32_bf16(false, a, false, b, (short)0, c,
                                                 false, false);
}

// A-fragment (16x32 bf16, row-major source):
// lanes 0-15 : M = lane,   K = kbase + {0..7, 16..23}
// lanes 16-31: M = lane-16, K = kbase + {8..15, 24..31}
static __device__ inline v16bf load_a_frag(const bf16* A, int lda, int row0,
                                           int kbase, int lane) {
  int l15 = lane & 15;
  int ko  = (lane & 16) ? 8 : 0;
  const bf16* p = A + (size_t)(row0 + l15) * lda + kbase + ko;
  v8bf lo = *(const v8bf*)(p);
  v8bf hi = *(const v8bf*)(p + 16);
  v16bf f;
#pragma unroll
  for (int u = 0; u < 8; ++u) { f[u] = lo[u]; f[u + 8] = hi[u]; }
  return f;
}

// B-fragment (32x16 bf16) from transposed-packed BT[N][K]:
// lane holds column n = tile + (lane&15), K = kbase + {0..7, 16..23}
// (+8 for the upper half-wave) -> two 16B loads.
static __device__ inline v16bf load_b_frag(const bf16* BT, int K, int ncol,
                                           int kbase, int lane) {
  int kb2 = kbase + ((lane & 16) ? 8 : 0);
  const bf16* p = BT + (size_t)ncol * K + kb2;
  v8bf lo = *(const v8bf*)(p);
  v8bf hi = *(const v8bf*)(p + 16);
  v16bf f;
#pragma unroll
  for (int u = 0; u < 8; ++u) { f[u] = lo[u]; f[u + 8] = hi[u]; }
  return f;
}

// ---------------------------------------------------------------- converters
__global__ void cvt_bf16_kernel(const float* in, bf16* out, int n) {
  int i = blockIdx.x * blockDim.x + threadIdx.x;
  if (i < n) out[i] = (bf16)in[i];
}

// out[n*K + k] = bf16(in[k*N + n]); grid.y = matrix (head) index
__global__ void packT_kernel(const float* in, bf16* out, int K, int N,
                             int inStride, int outStride) {
  int m = blockIdx.y;
  in  += (size_t)m * inStride;
  out += (size_t)m * outStride;
  int idx = blockIdx.x * blockDim.x + threadIdx.x;
  if (idx >= K * N) return;
  int n = idx / K, k = idx - n * K;
  out[idx] = (bf16)in[(size_t)k * N + n];
}

// ------------------------------------------------- generic WMMA GEMM (bf16)
// C(MxN, f32) = A(MxK, bf16 row-major) * BT(packed [N][K] bf16) [+ bias]
// block = 128 threads (4 waves), grid = (M/16, N/64); wave w -> n-tile.
__global__ void gemm_wmma_kernel(const bf16* A, const bf16* BT, const float* bias,
                                 float* C, int K, int N) {
  int tid  = threadIdx.x;
  int lane = tid & 31, wv = tid >> 5;
  int l15  = lane & 15, hiH = (lane >> 4) & 1;
  int mt = blockIdx.x * 16;
  int nt = (blockIdx.y * 4 + wv) * 16;

  v8f c;
#pragma unroll
  for (int r = 0; r < 8; ++r) c[r] = 0.0f;

  int nk = K >> 5;
  for (int ks = 0; ks < nk; ++ks) {
    v16bf af  = load_a_frag(A, K, mt, ks * 32, lane);
    v16bf bfr = load_b_frag(BT, K, nt + l15, ks * 32, lane);
    c = wmma_bf16(af, bfr, c);
  }
#pragma unroll
  for (int r = 0; r < 8; ++r) {
    int row = mt + r + 8 * hiH;
    int col = nt + l15;
    float bv = bias ? bias[col] : 0.0f;
    C[(size_t)row * N + col] = c[r] + bv;
  }
}

// ----------------------------------------------------- si / sjT (small, f32)
// si  = Wh @ Wa + ba1           (row-major   [h][b][j][e])
// sjT = (Wh @ Wb) transposed    (layout [h][b][e][j] for contiguous j reads)
__global__ void sisj_kernel(const float* Wh, const float* Wa1, const float* ba1,
                            float* si, float* sjT) {
  int row = blockIdx.x;            // 0 .. H*B*N-1 (layout [h][b][j])
  int hb  = row / Nx;              // hh*Bx + b
  int jn  = row - hb * Nx;
  int hh  = hb / Bx;
  int e   = threadIdx.x;           // 0..63
  const float* w  = Wh + (size_t)row * HDx;
  const float* Wa = Wa1 + (size_t)hh * 3 * HDx * HDx;   // rows 0..63
  const float* Wb = Wa + HDx * HDx;                     // rows 64..127
  float a = 0.f, bv = 0.f;
#pragma unroll 4
  for (int d = 0; d < HDx; ++d) {
    float x = w[d];
    a  += x * Wa[d * HDx + e];
    bv += x * Wb[d * HDx + e];
  }
  si[(size_t)row * HDx + e] = a + ba1[hh * HDx + e];
  sjT[((size_t)hb * HDx + e) * Nx + jn] = bv;
}

// ------------------------------------------- fused attention: one (b,h,i)
// edge GEMM (WMMA) -> hidden lrelu -> score dot -> softmax -> attn@Wh -> LN
__global__ void attn_kernel(const float* ei, const float* ej, const float* si,
                            const float* sjT, const bf16* WeT, const float* wa2,
                            const float* ba2, const int* adj, const float* Wh,
                            const float* ln_g, const float* ln_b, bf16* concat_bf) {
  int i = blockIdx.x, hh = blockIdx.y, b = blockIdx.z;
  int tid  = threadIdx.x;
  int lane = tid & 31, wv = tid >> 5;
  int l15  = lane & 15, hiH = (lane >> 4) & 1;

  __shared__ float ei_s[HDx], si_s[HDx], wa2_s[HDx], scores_s[Nx], hp_s[HDx];

  size_t headrow = ((size_t)hh * Bx + b) * Nx;   // row base for (h,b)
  const float* eib  = ei + (headrow + i) * HDx;
  const float* sib  = si + (headrow + i) * HDx;
  const float* ejb  = ej + headrow * HDx;
  const float* sjTb = sjT + ((size_t)hh * Bx + b) * HDx * Nx;
  const float* Whb  = Wh + headrow * HDx;
  const bf16*  WeTh = WeT + (size_t)hh * HDx * HDx;

  if (tid < HDx) {
    ei_s[tid]  = eib[tid];
    si_s[tid]  = sib[tid];
    wa2_s[tid] = wa2[hh * HDx + tid];
  }
  __syncthreads();

  // We (64x64) B-fragments are j-tile invariant: hoist all 8 into registers.
  v16bf bfrag[2][4];
#pragma unroll
  for (int ks = 0; ks < 2; ++ks)
#pragma unroll
    for (int nt = 0; nt < 4; ++nt)
      bfrag[ks][nt] = load_b_frag(WeTh, HDx, nt * 16 + l15, ks * 32, lane);

  int ko = hiH ? 8 : 0;

  // each wave owns j-tiles {wv, wv+4, wv+8, wv+12}
  for (int jt = wv; jt < Nx / 16; jt += 4) {
    int jrow = jt * 16 + l15;                 // A-fragment row (j index)
    const float* pj = ejb + (size_t)jrow * HDx;
    if (jt + 4 < Nx / 16) {                   // prefetch next tile's ej rows
      __builtin_prefetch(ejb + (size_t)(jrow + 64) * HDx, 0, 0);
    }

    v8f c[4];
    for (int q = 0; q < 4; ++q)
      for (int r = 0; r < 8; ++r) c[q][r] = 0.0f;

#pragma unroll
    for (int ks = 0; ks < 2; ++ks) {          // K = 64 -> 2 k-steps
      int kbase = ks * 32;
      v16bf af;
#pragma unroll
      for (int u = 0; u < 8; ++u) {           // edge = lrelu(ei + ej), bf16
        int k0 = kbase + ko + u, k1 = kbase + ko + 16 + u;
        float e0 = ei_s[k0] + pj[k0];
        float e1 = ei_s[k1] + pj[k1];
        e0 = e0 > 0.f ? e0 : ALPHA * e0;
        e1 = e1 > 0.f ? e1 : ALPHA * e1;
        af[u] = (bf16)e0; af[u + 8] = (bf16)e1;
      }
#pragma unroll
      for (int nt = 0; nt < 4; ++nt)          // N = 64 -> 4 n-tiles
        c[nt] = wmma_bf16(af, bfrag[ks][nt], c[nt]);
    }

    // C layout: lane = 16*half + n_local ; VGPR r -> M = r + 8*half (M = j)
    float s[8];
#pragma unroll
    for (int r = 0; r < 8; ++r) s[r] = 0.0f;
    int jloc = hiH * 8;
#pragma unroll
    for (int nt = 0; nt < 4; ++nt) {
      int n = nt * 16 + l15;
      const float* sp = sjTb + (size_t)n * Nx + jt * 16 + jloc;  // 8 contig
      float siv = si_s[n], w2 = wa2_s[n];
#pragma unroll
      for (int r = 0; r < 8; ++r) {
        float v = c[nt][r] + siv + sp[r];
        v = v > 0.f ? v : ALPHA * v;          // hidden = lrelu(...)
        s[r] += v * w2;                       // dot with wa2
      }
    }
    // sum over the 16 lanes of each half-wave (covers all 64 n)
#pragma unroll
    for (int m = 1; m < 16; m <<= 1)
#pragma unroll
      for (int r = 0; r < 8; ++r) s[r] += __shfl_xor(s[r], m, 32);
    if (l15 == 0) {
#pragma unroll
      for (int r = 0; r < 8; ++r) scores_s[jt * 16 + jloc + r] = s[r];
    }
  }
  __syncthreads();

  // wave 0: masked softmax over j (256 values, 8 per lane)
  if (wv == 0) {
    float ba2v = ba2[hh];
    const int* adjrow = adj + ((size_t)b * Nx + i) * Nx;
    float ev[8];
    float mx = -1e30f;
#pragma unroll
    for (int u = 0; u < 8; ++u) {
      int j = lane + u * 32;
      float e = scores_s[j] + ba2v;
      if (adjrow[j] == 0) e = -1.0e9f;
      ev[u] = e;
      mx = fmaxf(mx, e);
    }
#pragma unroll
    for (int m = 1; m < 32; m <<= 1) mx = fmaxf(mx, __shfl_xor(mx, m, 32));
    float sum = 0.f;
#pragma unroll
    for (int u = 0; u < 8; ++u) { ev[u] = __expf(ev[u] - mx); sum += ev[u]; }
#pragma unroll
    for (int m = 1; m < 32; m <<= 1) sum += __shfl_xor(sum, m, 32);
    float inv = 1.0f / sum;
#pragma unroll
    for (int u = 0; u < 8; ++u) scores_s[lane + u * 32] = ev[u] * inv;
  }
  __syncthreads();

  // hp[d] = sum_j attn[j] * Wh[j][d]   (threads 0..63, coalesced over d)
  if (tid < HDx) {
    float acc = 0.f;
    for (int j = 0; j < Nx; ++j) acc += scores_s[j] * Whb[(size_t)j * HDx + tid];
    hp_s[tid] = acc;
  }
  __syncthreads();

  // per-head layernorm over 64 dims, write bf16 concat slice
  if (tid < HDx) {
    float m = 0.f;
#pragma unroll 4
    for (int k = 0; k < HDx; ++k) m += hp_s[k];
    m *= (1.0f / HDx);
    float v = 0.f;
#pragma unroll 4
    for (int k = 0; k < HDx; ++k) { float t = hp_s[k] - m; v += t * t; }
    v *= (1.0f / HDx);
    float val = (hp_s[tid] - m) * rsqrtf(v + EPSLN) * ln_g[hh * HDx + tid]
              + ln_b[hh * HDx + tid];
    concat_bf[((size_t)b * Nx + i) * OUTx + hh * HDx + tid] = (bf16)val;
  }
}

// --------------------- output: concat @ Wo + bo + h, then layernorm(256)
// block = 128 (4 waves), grid = 128 m-tiles; wave w -> n-tiles 4w..4w+3
__global__ void out_ln_kernel(const bf16* A, const bf16* WoT, const float* bo,
                              const float* hres, const float* g2, const float* b2,
                              float* out) {
  __shared__ float tile[16 * OUTx];
  int tid  = threadIdx.x;
  int lane = tid & 31, wv = tid >> 5;
  int l15  = lane & 15, hiH = (lane >> 4) & 1;
  int mt = blockIdx.x * 16;

  v8f c[4];
  for (int q = 0; q < 4; ++q)
    for (int r = 0; r < 8; ++r) c[q][r] = 0.0f;

  for (int ks = 0; ks < 8; ++ks) {            // K = 256
    v16bf af = load_a_frag(A, OUTx, mt, ks * 32, lane);
#pragma unroll
    for (int q = 0; q < 4; ++q) {
      int n = (wv * 4 + q) * 16 + l15;
      v16bf bfr = load_b_frag(WoT, OUTx, n, ks * 32, lane);
      c[q] = wmma_bf16(af, bfr, c[q]);
    }
  }
#pragma unroll
  for (int q = 0; q < 4; ++q) {
    int col = (wv * 4 + q) * 16 + l15;
#pragma unroll
    for (int r = 0; r < 8; ++r) {
      int row = r + 8 * hiH;
      tile[row * OUTx + col] =
          c[q][r] + bo[col] + hres[(size_t)(mt + row) * OUTx + col];
    }
  }
  __syncthreads();

  if (tid < 16) {
    const float* rowp = tile + tid * OUTx;
    float m = 0.f;
    for (int k = 0; k < OUTx; ++k) m += rowp[k];
    m *= (1.0f / OUTx);
    float v = 0.f;
    for (int k = 0; k < OUTx; ++k) { float t = rowp[k] - m; v += t * t; }
    v *= (1.0f / OUTx);
    float is = rsqrtf(v + EPSLN);
    float* op = out + (size_t)(mt + tid) * OUTx;
    for (int k = 0; k < OUTx; ++k)
      op[k] = (rowp[k] - m) * is * g2[k] + b2[k];
  }
}

// --------------------------------------------------------------------------
extern "C" void kernel_launch(void* const* d_in, const int* in_sizes, int n_in,
                              void* d_out, int out_size, void* d_ws, size_t ws_size,
                              hipStream_t stream) {
  const float* h    = (const float*)d_in[0];
  const int*   adj  = (const int*)d_in[1];
  const float* W    = (const float*)d_in[2];
  const float* We1  = (const float*)d_in[3];
  const float* be1  = (const float*)d_in[4];
  const float* Wa1  = (const float*)d_in[5];
  const float* ba1  = (const float*)d_in[6];
  const float* wa2  = (const float*)d_in[7];
  const float* ba2  = (const float*)d_in[8];
  const float* ln_g = (const float*)d_in[9];
  const float* ln_b = (const float*)d_in[10];
  const float* Wo   = (const float*)d_in[11];
  const float* bo   = (const float*)d_in[12];
  const float* g2   = (const float*)d_in[13];
  const float* b2   = (const float*)d_in[14];
  float* out = (float*)d_out;

  const size_t ROWS = (size_t)Bx * Nx;           // 2048
  const size_t HB   = (size_t)Hx * ROWS * HDx;   // per-head-array elem count

  char* p = (char*)d_ws;
  auto take = [&](size_t bytes) {
    char* q = p;
    p += (bytes + 255) & ~(size_t)255;
    return q;
  };
  bf16* h_bf      = (bf16*)take(ROWS * INx * 2);
  bf16* WT_W      = (bf16*)take((size_t)Hx * HDx * INx * 2);
  bf16* WT_Ea     = (bf16*)take((size_t)Hx * HDx * INx * 2);
  bf16* WT_Eb     = (bf16*)take((size_t)Hx * HDx * INx * 2);
  bf16* WeT       = (bf16*)take((size_t)Hx * HDx * HDx * 2);
  bf16* WoT       = (bf16*)take((size_t)OUTx * OUTx * 2);
  float* Wh_g     = (float*)take(HB * 4);
  float* ei_g     = (float*)take(HB * 4);
  float* ej_g     = (float*)take(HB * 4);
  float* si_g     = (float*)take(HB * 4);
  float* sjT_g    = (float*)take(HB * 4);
  bf16* concat_bf = (bf16*)take(ROWS * OUTx * 2);

  // 1) input conversion + transposed bf16 weight packs
  cvt_bf16_kernel<<<dim3((int)(ROWS * INx / 256)), 256, 0, stream>>>(
      h, h_bf, (int)(ROWS * INx));
  packT_kernel<<<dim3((INx * HDx + 255) / 256, Hx), 256, 0, stream>>>(
      W, WT_W, INx, HDx, INx * HDx, HDx * INx);
  packT_kernel<<<dim3((INx * HDx + 255) / 256, Hx), 256, 0, stream>>>(
      We1, WT_Ea, INx, HDx, 2 * INx * HDx, HDx * INx);
  packT_kernel<<<dim3((INx * HDx + 255) / 256, Hx), 256, 0, stream>>>(
      We1 + (size_t)INx * HDx, WT_Eb, INx, HDx, 2 * INx * HDx, HDx * INx);
  packT_kernel<<<dim3((HDx * HDx + 255) / 256, Hx), 256, 0, stream>>>(
      Wa1 + 2 * HDx * HDx, WeT, HDx, HDx, 3 * HDx * HDx, HDx * HDx);
  packT_kernel<<<dim3((OUTx * OUTx + 255) / 256, 1), 256, 0, stream>>>(
      Wo, WoT, OUTx, OUTx, 0, 0);

  // 2) projection GEMMs per head: Wh, ei(+be1), ej
  for (int hh = 0; hh < Hx; ++hh) {
    size_t wofs = (size_t)hh * HDx * INx;
    size_t cofs = (size_t)hh * ROWS * HDx;
    gemm_wmma_kernel<<<dim3((int)(ROWS / 16), 1), 128, 0, stream>>>(
        h_bf, WT_W + wofs, nullptr, Wh_g + cofs, INx, HDx);
    gemm_wmma_kernel<<<dim3((int)(ROWS / 16), 1), 128, 0, stream>>>(
        h_bf, WT_Ea + wofs, be1 + hh * HDx, ei_g + cofs, INx, HDx);
    gemm_wmma_kernel<<<dim3((int)(ROWS / 16), 1), 128, 0, stream>>>(
        h_bf, WT_Eb + wofs, nullptr, ej_g + cofs, INx, HDx);
  }

  // 3) si / sjT (small fp32)
  sisj_kernel<<<dim3(Hx * Bx * Nx), 64, 0, stream>>>(Wh_g, Wa1, ba1, si_g, sjT_g);

  // 4) fused attention per (b, head, i)
  attn_kernel<<<dim3(Nx, Hx, Bx), 128, 0, stream>>>(
      ei_g, ej_g, si_g, sjT_g, WeT, wa2, ba2, adj, Wh_g, ln_g, ln_b, concat_bf);

  // 5) output GEMM + bias + residual + final layernorm
  out_ln_kernel<<<dim3((int)(ROWS / 16)), 128, 0, stream>>>(
      concat_bf, WoT, bo, h, g2, b2, out);
}
